// VSSOnewayLayer_3238405341501
// MI455X (gfx1250) — matrix operand order repaired and checked by reading, not compiled, not run on hardware
//
#include <hip/hip_runtime.h>
#include <hip/hip_bf16.h>

// ---------------- problem constants ----------------
#define B_   8
#define L_   2048
#define D_   256
#define DIN_ 512
#define N_   64
#define P_   64
#define H_   8
#define NPAD_ 144           // H_ + 2*N_ = 136 padded to 9 WMMA tiles
#define BL_  (B_ * L_)      // 16384 rows

typedef __attribute__((ext_vector_type(16))) _Float16 v16h;
typedef __attribute__((ext_vector_type(8)))  _Float16 v8h;
typedef __attribute__((ext_vector_type(8)))  float    v8f;

// ---------------------------------------------------------------------------
// WMMA fragment loader: two contiguous 8xf16 (16-byte) runs -> one v16h frag
// ---------------------------------------------------------------------------
__device__ __forceinline__ v16h make_frag(const _Float16* p0, const _Float16* p1) {
    v8h lo = *(const v8h*)p0;
    v8h hi = *(const v8h*)p1;
    v16h r;
#pragma unroll
    for (int i = 0; i < 8; ++i) { r[i] = lo[i]; r[i + 8] = hi[i]; }
    return r;
}

// ---------------------------------------------------------------------------
// f16 WMMA GEMM with (TM x TN) 16x16-tile macro-tile per wave:
//   C[M x N] = A[M x K] * BT[N x K]^T  (+ optional residual)
// A row-major (K contiguous), BT row-major (K contiguous) = B transposed.
// Per 32-K step: TM+TN fragment loads feed TM*TN v_wmma ops (2x2 -> 1 load/wmma).
// ---------------------------------------------------------------------------
template <int TM, int TN>
__global__ void wmma_gemm_f16(const _Float16* __restrict__ A,
                              const _Float16* __restrict__ BT,
                              float* __restrict__ C,
                              const float* __restrict__ res,
                              int M, int N, int K, int ldc) {
    const int wave = threadIdx.x >> 5;
    const int lane = threadIdx.x & 31;
    const int nt = (N >> 4) / TN;                  // macro-cols
    const int mt = (M >> 4) / TM;                  // macro-rows
    const int tile = blockIdx.x * (blockDim.x >> 5) + wave;
    if (tile >= mt * nt) return;                   // wave-uniform: EXEC stays all-ones

    const int tm = tile / nt;
    const int tn = tile - tm * nt;
    const int row = lane & 15;                     // A row / B column within 16-tile
    const int hi  = lane >> 4;

    const _Float16* Arow[TM];
    const _Float16* Bcol[TN];
#pragma unroll
    for (int i = 0; i < TM; ++i)
        Arow[i] = A + (size_t)((tm * TM + i) * 16 + row) * K;
#pragma unroll
    for (int j = 0; j < TN; ++j)
        Bcol[j] = BT + (size_t)((tn * TN + j) * 16 + row) * K;

    v8f acc[TM][TN] = {};
    for (int k0 = 0; k0 < K; k0 += 32) {
        v16h a[TM], b[TN];
#pragma unroll
        for (int i = 0; i < TM; ++i)
            a[i] = make_frag(Arow[i] + k0 + hi * 8, Arow[i] + k0 + 16 + hi * 8);
#pragma unroll
        for (int j = 0; j < TN; ++j)
            b[j] = make_frag(Bcol[j] + k0 + hi * 8, Bcol[j] + k0 + 16 + hi * 8);
#pragma unroll
        for (int i = 0; i < TM; ++i)
#pragma unroll
            for (int j = 0; j < TN; ++j)
                acc[i][j] = __builtin_amdgcn_wmma_f32_16x16x32_f16(
                    false, a[i], false, b[j], (short)0, acc[i][j], false, false);
    }

#pragma unroll
    for (int i = 0; i < TM; ++i) {
#pragma unroll
        for (int j = 0; j < TN; ++j) {
            const int col = (tn * TN + j) * 16 + (lane & 15);
#pragma unroll
            for (int r = 0; r < 8; ++r) {
                size_t o = (size_t)((tm * TM + i) * 16 + hi * 8 + r) * ldc + col;
                float v = acc[i][j][r];
                if (res) v += res[o];
                C[o] = v;
            }
        }
    }
}

// ---------------------------------------------------------------------------
// LayerNorm over last dim D (256 or 512), fp32 in -> f16 out (WMMA A operand)
// ---------------------------------------------------------------------------
template <int D>
__global__ void ln_f16_kernel(const float* __restrict__ in,
                              const float* __restrict__ gam,
                              const float* __restrict__ bet,
                              _Float16* __restrict__ out) {
    constexpr int T = 256;
    constexpr int E = D / T;
    __shared__ float s1[T], s2[T];
    const float* row = in + (size_t)blockIdx.x * D;
    float v[E], sum = 0.f, sq = 0.f;
#pragma unroll
    for (int e = 0; e < E; ++e) {
        v[e] = row[threadIdx.x + e * T];
        sum += v[e]; sq += v[e] * v[e];
    }
    s1[threadIdx.x] = sum; s2[threadIdx.x] = sq;
    __syncthreads();
    for (int off = T / 2; off > 0; off >>= 1) {
        if (threadIdx.x < off) {
            s1[threadIdx.x] += s1[threadIdx.x + off];
            s2[threadIdx.x] += s2[threadIdx.x + off];
        }
        __syncthreads();
    }
    const float mean = s1[0] * (1.f / D);
    const float var  = s2[0] * (1.f / D) - mean * mean;
    const float rs   = rsqrtf(var + 1e-5f);
    _Float16* orow = out + (size_t)blockIdx.x * D;
#pragma unroll
    for (int e = 0; e < E; ++e) {
        int d = threadIdx.x + e * T;
        orow[d] = (_Float16)((v[e] - mean) * rs * gam[d] + bet[d]);
    }
}

// ---------------------------------------------------------------------------
// Weight transpose + f32->f16 convert (+ zero pad): out[n][k] = in[k][n]
// ---------------------------------------------------------------------------
__global__ void transpose_f16_kernel(const float* __restrict__ in,
                                     _Float16* __restrict__ out,
                                     int K, int Nsrc, int Npad) {
    int idx = blockIdx.x * blockDim.x + threadIdx.x;
    if (idx >= K * Npad) return;
    int n = idx / K, k = idx - n * K;
    float v = (n < Nsrc) ? in[(size_t)k * Nsrc + n] : 0.f;
    out[(size_t)n * K + k] = (_Float16)v;
}

// ---------------------------------------------------------------------------
// Depthwise conv3 (SAME, per-L) + bias + exact GELU; emits f32 + f16 copies
// ---------------------------------------------------------------------------
__global__ void conv_gelu_kernel(const float* __restrict__ up,
                                 const float* __restrict__ cw,
                                 const float* __restrict__ cb,
                                 float* __restrict__ u32,
                                 _Float16* __restrict__ u16) {
    size_t idx = (size_t)blockIdx.x * blockDim.x + threadIdx.x;
    if (idx >= (size_t)BL_ * DIN_) return;
    int c = (int)(idx & (DIN_ - 1));
    size_t bl = idx >> 9;                 // / DIN_
    int l = (int)(bl & (L_ - 1));
    float w0 = cw[c * 3 + 0], w1 = cw[c * 3 + 1], w2 = cw[c * 3 + 2];
    float xm = (l > 0)      ? up[idx - DIN_] : 0.f;
    float x0 = up[idx];
    float xp = (l < L_ - 1) ? up[idx + DIN_] : 0.f;
    float v = xm * w0 + x0 * w1 + xp * w2 + cb[c];
    v = 0.5f * v * (1.f + erff(v * 0.70710678118654752f));   // exact GELU
    u32[idx] = v;
    u16[idx] = (_Float16)v;
}

// ---------------------------------------------------------------------------
// Selective scan. One block per (b,h); state h[P=64][N=64] lives in registers:
// thread t owns (p = t>>2, n = (t&3)*16 .. +15)  => 16 f32 regs each.
// y contraction over N reduced across the 4 owning lanes via wave32 shfl_xor.
// global_prefetch_b8 issued 8 steps ahead to hide L2 latency behind recurrence.
// ---------------------------------------------------------------------------
__global__ void scan_kernel(const float* __restrict__ dbc,   // [BL][NPAD_]
                            const float* __restrict__ u,     // [BL][DIN_] f32
                            const float* __restrict__ dt_bias,
                            const float* __restrict__ A_log,
                            const float* __restrict__ Ds,
                            float* __restrict__ y) {         // [BL][DIN_]
    const int bh = blockIdx.x;
    const int b  = bh >> 3;            // H_ = 8
    const int h  = bh & 7;
    const int t  = threadIdx.x;
    const int p  = t >> 2;
    const int nseg = (t & 3) << 4;

    float st[16];
#pragma unroll
    for (int i = 0; i < 16; ++i) st[i] = 0.f;

    const float Ah   = -__expf(A_log[h]);
    const float bias = dt_bias[h];
    const float Dh   = Ds[h];
    const size_t base = (size_t)b * L_;

    for (int l = 0; l < L_; ++l) {
        // ---- prefetch 8 steps ahead (L2-resident working set) ----
        if (l + 8 < L_) {
            __builtin_prefetch(dbc + (base + l + 8) * NPAD_ + H_ + nseg, 0, 0);
            __builtin_prefetch(u + (base + l + 8) * DIN_ + h * P_ + p, 0, 0);
        }

        const float* row = dbc + (base + l) * NPAD_;
        float dr  = row[h] + bias;
        float dt  = (dr > 20.f) ? dr : log1pf(__expf(dr));     // softplus
        float dec = __expf(dt * Ah);
        float xp  = u[(base + l) * DIN_ + h * P_ + p];
        float dx  = dt * xp;

        const float4* Bv = (const float4*)(row + H_ + nseg);        // B block
        const float4* Cv = (const float4*)(row + H_ + N_ + nseg);   // C block
        float acc = 0.f;
#pragma unroll
        for (int i = 0; i < 4; ++i) {
            float4 Bq = Bv[i], Cq = Cv[i];
            st[4*i+0] = st[4*i+0] * dec + dx * Bq.x; acc += st[4*i+0] * Cq.x;
            st[4*i+1] = st[4*i+1] * dec + dx * Bq.y; acc += st[4*i+1] * Cq.y;
            st[4*i+2] = st[4*i+2] * dec + dx * Bq.z; acc += st[4*i+2] * Cq.z;
            st[4*i+3] = st[4*i+3] * dec + dx * Bq.w; acc += st[4*i+3] * Cq.w;
        }
        acc += __shfl_xor(acc, 1, 32);
        acc += __shfl_xor(acc, 2, 32);
        if ((t & 3) == 0)
            y[(base + l) * DIN_ + h * P_ + p] = acc + xp * Dh;
    }
}

// ---------------------------------------------------------------------------
extern "C" void kernel_launch(void* const* d_in, const int* in_sizes, int n_in,
                              void* d_out, int out_size, void* d_ws, size_t ws_size,
                              hipStream_t stream) {
    const float* src     = (const float*)d_in[0];
    const float* ln_w    = (const float*)d_in[1];
    const float* ln_b    = (const float*)d_in[2];
    const float* W_in    = (const float*)d_in[3];
    const float* conv_w  = (const float*)d_in[4];
    const float* conv_b  = (const float*)d_in[5];
    const float* W_xproj = (const float*)d_in[6];
    const float* dt_bias = (const float*)d_in[7];
    const float* A_log   = (const float*)d_in[8];
    const float* Ds      = (const float*)d_in[9];
    const float* oln_w   = (const float*)d_in[10];
    const float* oln_b   = (const float*)d_in[11];
    const float* W_out   = (const float*)d_in[12];
    float* out = (float*)d_out;

    // ---- workspace partition (all slices multiples of 256 bytes) ----
    char* w = (char*)d_ws;
    _Float16* x16   = (_Float16*)w; w += (size_t)BL_ * D_ * 2;        // 8 MB
    _Float16* WinT  = (_Float16*)w; w += (size_t)DIN_ * D_ * 2;       // 256 KB  [512][256]
    _Float16* WxT   = (_Float16*)w; w += (size_t)NPAD_ * DIN_ * 2;    // 144 KB  [144][512]
    _Float16* WoutT = (_Float16*)w; w += (size_t)D_ * DIN_ * 2;       // 256 KB  [256][512]
    float*    u_pre = (float*)w;    w += (size_t)BL_ * DIN_ * 4;      // 32 MB
    float*    u32   = (float*)w;    w += (size_t)BL_ * DIN_ * 4;      // 32 MB
    _Float16* u16   = (_Float16*)w; w += (size_t)BL_ * DIN_ * 2;      // 16 MB
    float*    dbc   = (float*)w;    w += (size_t)BL_ * NPAD_ * 4;     // 9.4 MB
    float*    yss   = (float*)w;    w += (size_t)BL_ * DIN_ * 4;      // 32 MB
    _Float16* yln   = (_Float16*)w; w += (size_t)BL_ * DIN_ * 2;      // 16 MB

    // ---- one-time weight transposes (f32 -> f16, K-contiguous) ----
    transpose_f16_kernel<<<(D_ * DIN_ + 255) / 256, 256, 0, stream>>>(
        W_in, WinT, /*K=*/D_, /*Nsrc=*/DIN_, /*Npad=*/DIN_);
    transpose_f16_kernel<<<(DIN_ * NPAD_ + 255) / 256, 256, 0, stream>>>(
        W_xproj, WxT, /*K=*/DIN_, /*Nsrc=*/H_ + 2 * N_, /*Npad=*/NPAD_);
    transpose_f16_kernel<<<(DIN_ * D_ + 255) / 256, 256, 0, stream>>>(
        W_out, WoutT, /*K=*/DIN_, /*Nsrc=*/D_, /*Npad=*/D_);

    // ---- 1) input LN -> f16 ----
    ln_f16_kernel<D_><<<BL_, 256, 0, stream>>>(src, ln_w, ln_b, x16);

    // ---- 2) u_pre = x @ W_in  (16384x512, K=256), 32x32 macro-tiles ----
    {
        int waves = (BL_ / 32) * (DIN_ / 32);
        wmma_gemm_f16<2, 2><<<(waves + 7) / 8, 256, 0, stream>>>(
            x16, WinT, u_pre, nullptr, BL_, DIN_, D_, DIN_);
    }

    // ---- 3) depthwise conv3 + bias + GELU ----
    conv_gelu_kernel<<<(BL_ * DIN_ + 255) / 256, 256, 0, stream>>>(
        u_pre, conv_w, conv_b, u32, u16);

    // ---- 4) dbc = u @ W_xproj  (16384x144pad, K=512), 32x16 macro-tiles ----
    {
        int waves = (BL_ / 32) * (NPAD_ / 16);
        wmma_gemm_f16<2, 1><<<(waves + 7) / 8, 256, 0, stream>>>(
            u16, WxT, dbc, nullptr, BL_, NPAD_, DIN_, NPAD_);
    }

    // ---- 5) selective scan (+ Ds skip) ----
    scan_kernel<<<B_ * H_, 256, 0, stream>>>(dbc, u32, dt_bias, A_log, Ds, yss);

    // ---- 6) output LN -> f16 ----
    ln_f16_kernel<DIN_><<<BL_, 256, 0, stream>>>(yss, oln_w, oln_b, yln);

    // ---- 7) out = src + yln @ W_out  (16384x256, K=512), 32x32 macro-tiles ----
    {
        int waves = (BL_ / 32) * (D_ / 32);
        wmma_gemm_f16<2, 2><<<(waves + 7) / 8, 256, 0, stream>>>(
            yln, WoutT, out, src, BL_, D_, DIN_, D_);
    }
    (void)in_sizes; (void)n_in; (void)out_size; (void)ws_size;
}
